// CSINet_37082747634498
// MI455X (gfx1250) — matrix operand order repaired and checked by reading, not compiled, not run on hardware
//
#include <hip/hip_runtime.h>
#include <hip/hip_bf16.h>
#include <math.h>

typedef __attribute__((ext_vector_type(16))) _Float16 v16h;
typedef __attribute__((ext_vector_type(8)))  float    v8f;

#define NN   32
#define PP   992
#define DD   256
#define MMs  14
#define OCC  151
#define RCC  51
#define K1   4251
#define K1P  4256

union HFrag { v16h v; float4 f4[2]; };

// ---------------------------------------------------------------------------
// Weight convert: f32 (K x N) row-major  ->  f16 transposed (Np x Kp), zero pad
// ---------------------------------------------------------------------------
__global__ void wt_transpose_f16(const float* __restrict__ w, _Float16* __restrict__ o,
                                 int K, int N, int Kp, int Np) {
  int idx = blockIdx.x * 256 + threadIdx.x;
  if (idx >= Np * Kp) return;
  int n = idx / Kp, k = idx % Kp;
  float v = (k < K && n < N) ? w[k * N + n] : 0.f;
  o[idx] = (_Float16)v;
}

// ---------------------------------------------------------------------------
// Build obj_in = concat([roi(4096) | logits(151) | bbox(4) | pad]) in f16
// ---------------------------------------------------------------------------
__global__ void build_objin16(const float* __restrict__ roi, const float* __restrict__ logits,
                              const float* __restrict__ bbox, _Float16* __restrict__ out) {
  int idx = blockIdx.x * 256 + threadIdx.x;
  if (idx >= NN * K1P) return;
  int m = idx / K1P, k = idx % K1P;
  float v;
  if      (k < 4096)       v = roi[m * 4096 + k];
  else if (k < 4096 + OCC) v = logits[m * OCC + (k - 4096)];
  else if (k < K1)         v = bbox[m * 4 + (k - 4096 - OCC)];
  else                     v = 0.f;
  out[idx] = (_Float16)v;
}

// ---------------------------------------------------------------------------
// Masked spatial means over union_features: the 199MB streaming pass.
// One block per pair; masks (subject/object/background) -> LDS float4;
// thread c streams its 196-float channel with float4 loads.
// ---------------------------------------------------------------------------
__global__ void masked_mean_kernel(const float* __restrict__ uf,
                                   const float* __restrict__ bboxes,
                                   const int*   __restrict__ pairs,
                                   float* __restrict__ sF32, _Float16* __restrict__ s16) {
  __shared__ float4 wm[196];
  int p = blockIdx.x;
  int t = threadIdx.x;
  int si = pairs[2 * p], oi = pairs[2 * p + 1];
  if (t < 196) {
    float sx0 = bboxes[si*4+0], sy0 = bboxes[si*4+1], sx1 = bboxes[si*4+2], sy1 = bboxes[si*4+3];
    float ox0 = bboxes[oi*4+0], oy0 = bboxes[oi*4+1], ox1 = bboxes[oi*4+2], oy1 = bboxes[oi*4+3];
    float ux = fminf(sx0, ox0), uy = fminf(sy0, oy0);
    float xs0 = sx0-ux, xs1 = sx1-ux, xo0 = ox0-ux, xo1 = ox1-ux;
    float ys0 = sy0-uy, ys1 = sy1-uy, yo0 = oy0-uy, yo1 = oy1-uy;
    float xr = (float)MMs / fmaxf(xs1, xo1);
    float yr = (float)MMs / fmaxf(ys1, yo1);
    // jnp.round is round-half-even -> rintf under default RNE
    float xps0 = rintf(xs0*xr), xps1 = rintf(xs1*xr), xpo0 = rintf(xo0*xr), xpo1 = rintf(xo1*xr);
    float yps0 = rintf(ys0*yr), yps1 = rintf(ys1*yr), ypo0 = rintf(yo0*yr), ypo1 = rintf(yo1*yr);
    float fi = (float)(t / MMs), fj = (float)(t % MMs);
    float fS = (fi >= xps0 && fi < xps1 && fj >= yps0 && fj < yps1) ? 1.f : 0.f;
    float fO = (fi >= xpo0 && fi < xpo1 && fj >= ypo0 && fj < ypo1) ? 1.f : 0.f;
    float fB = fminf(fmaxf(1.f - fS - fO, 0.f), 1.f);
    wm[t] = make_float4(fS, fO, fB, 0.f);
  }
  __syncthreads();
  int c = t;  // 256 channels
  const float4* up = reinterpret_cast<const float4*>(uf + ((size_t)p * DD + c) * 196);
  float aS = 0.f, aO = 0.f, aB = 0.f;
  #pragma unroll 7
  for (int k = 0; k < 49; ++k) {
    float4 u  = up[k];
    float4 w0 = wm[4*k+0], w1 = wm[4*k+1], w2 = wm[4*k+2], w3 = wm[4*k+3];
    aS += u.x*w0.x + u.y*w1.x + u.z*w2.x + u.w*w3.x;
    aO += u.x*w0.y + u.y*w1.y + u.z*w2.y + u.w*w3.y;
    aB += u.x*w0.z + u.y*w1.z + u.z*w2.z + u.w*w3.z;
  }
  const float inv = 1.f / 196.f;
  size_t base = (size_t)p * DD + c;
  size_t PD = (size_t)PP * DD;
  float vS = aS*inv, vO = aO*inv, vB = aB*inv;
  sF32[base] = vS;  sF32[PD + base] = vO;  sF32[2*PD + base] = vB;
  s16 [base] = (_Float16)vS;  s16[PD + base] = (_Float16)vO;  s16[2*PD + base] = (_Float16)vB;
}

// ---------------------------------------------------------------------------
// WMMA GEMM: C(MxN) = A(MxK,f16) * B(KxN)  [B given transposed: Bt (Npad x K), f16]
// One 16x16 tile per wave, 4 waves/block. f32 accumulate via v_wmma_f32_16x16x32_f16.
// Optional bias / relu; writes f32 and/or f16 outputs (ldc = Nreal).
// ---------------------------------------------------------------------------
__global__ void gemm_wmma_f16(const _Float16* __restrict__ A,
                              const _Float16* __restrict__ Bt,
                              const float* __restrict__ bias,
                              float* __restrict__ Cf,
                              _Float16* __restrict__ Ch,
                              int ldc, int Nreal, int Ntiles, int K, int relu) {
  int ntile = blockIdx.y * 4 + threadIdx.y;
  if (ntile >= Ntiles) return;           // whole-wave uniform exit; EXEC stays all-ones
  int mtile = blockIdx.x;
  int lane = threadIdx.x;
  int l16 = lane & 15;
  int hi  = lane >> 4;
  const _Float16* arow = A  + (size_t)(mtile * 16 + l16) * K;
  const _Float16* brow = Bt + (size_t)(ntile * 16 + l16) * K;
  v8f c = {0.f, 0.f, 0.f, 0.f, 0.f, 0.f, 0.f, 0.f};
  for (int k0 = 0; k0 < K; k0 += 32) {
    HFrag a, b;
    // A fragment: lanes<16 rows M=l16 hold K {0..7,16..23}; lanes>=16 hold {8..15,24..31}
    a.f4[0] = *reinterpret_cast<const float4*>(arow + k0 + hi * 8);
    a.f4[1] = *reinterpret_cast<const float4*>(arow + k0 + hi * 8 + 16);
    // B fragment: lanes<16 col N=l16 hold K {0..7,16..23}; lanes>=16 hold {8..15,24..31}
    const _Float16* bp = brow + k0 + hi * 8;
    b.f4[0] = *reinterpret_cast<const float4*>(bp);
    b.f4[1] = *reinterpret_cast<const float4*>(bp + 16);
    c = __builtin_amdgcn_wmma_f32_16x16x32_f16(false, a.v, false, b.v, (short)0, c,
                                               false, false);
  }
  int col = ntile * 16 + l16;
  if (col < Nreal) {
    float bv = bias ? bias[col] : 0.f;
    #pragma unroll
    for (int r = 0; r < 8; ++r) {
      int row = mtile * 16 + hi * 8 + r;   // VGPR r: lanes<16 -> M=r, lanes>=16 -> M=8+r
      float v = c[r] + bv;
      if (relu) v = fmaxf(v, 0.f);
      if (Cf) Cf[(size_t)row * ldc + col] = v;
      if (Ch) Ch[(size_t)row * ldc + col] = (_Float16)v;
    }
  }
}

// ---------------------------------------------------------------------------
// Attention gate: rel16[p, t*256+c] = s[t,p,c] * sigmoid(pre[p,c])
// ---------------------------------------------------------------------------
__global__ void gate_kernel(const float* __restrict__ sF32, const float* __restrict__ pre,
                            _Float16* __restrict__ rel16, int t) {
  int p = blockIdx.x, c = threadIdx.x;
  float s = sF32[(size_t)t * PP * DD + (size_t)p * DD + c];
  float x = pre[(size_t)p * DD + c];
  float g = 1.f / (1.f + expf(-x));
  rel16[(size_t)p * (3 * DD) + t * DD + c] = (_Float16)(s * g);
}

// ---------------------------------------------------------------------------
// GCN combine (structural sparse norm_adj @ h), object rows.
// ---------------------------------------------------------------------------
__global__ void gcn_obj_kernel(const float* __restrict__ h, const float* __restrict__ feats,
                               const float* __restrict__ bias, const int* __restrict__ pairs,
                               _Float16* __restrict__ out16) {
  int i = blockIdx.x, c = threadIdx.x;
  const int2* pr = reinterpret_cast<const int2*>(pairs);
  float acc = h[(size_t)i * DD + c];   // eye (diag) term
  int deg = 1;
  for (int r = 0; r < PP; ++r) {
    int2 so = pr[r];
    if (so.x == i) { acc += h[(size_t)so.y * DD + c]; deg++; }          // o2o edge
    if (so.x == i || so.y == i) { acc += h[(size_t)(NN + r) * DD + c]; deg++; }  // o2r edge
  }
  float v = fmaxf(acc / (float)deg + bias[c], 0.f) + feats[(size_t)i * DD + c];
  out16[(size_t)i * DD + c] = (_Float16)v;
}

// relation rows: (h_rel[r] + h_obj[s] + h_obj[o]) / 3, relu(+bias) + residual
__global__ void gcn_rel_kernel(const float* __restrict__ h, const float* __restrict__ feats,
                               const float* __restrict__ bias, const int* __restrict__ pairs,
                               _Float16* __restrict__ out16) {
  int r = blockIdx.x, c = threadIdx.x;
  int s = pairs[2 * r], o = pairs[2 * r + 1];
  float acc = h[(size_t)(NN + r) * DD + c] + h[(size_t)s * DD + c] + h[(size_t)o * DD + c];
  float deg = (s == o) ? 2.f : 3.f;
  float v = fmaxf(acc / deg + bias[c], 0.f) + feats[(size_t)(NN + r) * DD + c];
  out16[(size_t)(NN + r) * DD + c] = (_Float16)v;
}

// ---------------------------------------------------------------------------
static inline char* bump(char*& p, size_t bytes) {
  char* r = p;
  p += (bytes + 255) & ~(size_t)255;
  return r;
}

extern "C" void kernel_launch(void* const* d_in, const int* in_sizes, int n_in,
                              void* d_out, int out_size, void* d_ws, size_t ws_size,
                              hipStream_t stream) {
  const float* roi       = (const float*)d_in[0];
  const float* bbox      = (const float*)d_in[1];
  const float* logits    = (const float*)d_in[2];
  const float* uf        = (const float*)d_in[3];
  const int*   pairs     = (const int*)  d_in[4];
  const float* W_emb1    = (const float*)d_in[5];
  const float* b_emb1    = (const float*)d_in[6];
  const float* W_emb2    = (const float*)d_in[7];
  const float* b_emb2    = (const float*)d_in[8];
  const float* att1_w    = (const float*)d_in[9];
  const float* att1_b    = (const float*)d_in[10];
  const float* att2_w    = (const float*)d_in[11];
  const float* att2_b    = (const float*)d_in[12];
  const float* W_c1      = (const float*)d_in[13];
  const float* b_c1      = (const float*)d_in[14];
  const float* W_c2      = (const float*)d_in[15];
  const float* b_c2      = (const float*)d_in[16];
  const float* W_gcn     = (const float*)d_in[17];
  const float* b_gcn     = (const float*)d_in[18];
  const float* W_objp    = (const float*)d_in[19];
  const float* b_objp    = (const float*)d_in[20];
  const float* W_relp    = (const float*)d_in[21];
  const float* b_relp    = (const float*)d_in[22];
  float* out = (float*)d_out;

  char* wp = (char*)d_ws;
  _Float16* wemb1T = (_Float16*)bump(wp, (size_t)256 * K1P * 2);
  _Float16* wemb2T = (_Float16*)bump(wp, (size_t)256 * 256 * 2);
  _Float16* wc1T   = (_Float16*)bump(wp, (size_t)256 * 768 * 2);
  _Float16* wc2T   = (_Float16*)bump(wp, (size_t)256 * 256 * 2);
  _Float16* wgcnT  = (_Float16*)bump(wp, (size_t)256 * 256 * 2);
  _Float16* wobjpT = (_Float16*)bump(wp, (size_t)160 * 256 * 2);
  _Float16* wrelpT = (_Float16*)bump(wp, (size_t)64 * 256 * 2);
  _Float16* watt1T = (_Float16*)bump(wp, (size_t)3 * 64 * 256 * 2);   // per-t (64 x 256)
  _Float16* watt2T = (_Float16*)bump(wp, (size_t)3 * 256 * 64 * 2);   // per-t (256 x 64)
  _Float16* objin16= (_Float16*)bump(wp, (size_t)NN * K1P * 2);
  float*    sF32   = (float*)   bump(wp, (size_t)3 * PP * DD * 4);
  _Float16* s16    = (_Float16*)bump(wp, (size_t)3 * PP * DD * 2);
  _Float16* g1h    = (_Float16*)bump(wp, (size_t)PP * 64 * 2);
  float*    pre    = (float*)   bump(wp, (size_t)PP * DD * 4);
  _Float16* rel16  = (_Float16*)bump(wp, (size_t)PP * 768 * 2);
  _Float16* h1o16  = (_Float16*)bump(wp, (size_t)NN * DD * 2);
  _Float16* h1r16  = (_Float16*)bump(wp, (size_t)PP * DD * 2);
  float*    feats  = (float*)   bump(wp, (size_t)(NN + PP) * DD * 4);
  _Float16* feats16= (_Float16*)bump(wp, (size_t)(NN + PP) * DD * 2);
  float*    hbuf   = (float*)   bump(wp, (size_t)(NN + PP) * DD * 4);
  _Float16* gcn16  = (_Float16*)bump(wp, (size_t)(NN + PP) * DD * 2);

  dim3 wblk(32, 4);

  // --- weight transposes / f16 conversion ---
  hipLaunchKernelGGL(wt_transpose_f16, dim3(4256), dim3(256), 0, stream, W_emb1, wemb1T, K1, 256, K1P, 256);
  hipLaunchKernelGGL(wt_transpose_f16, dim3(256),  dim3(256), 0, stream, W_emb2, wemb2T, 256, 256, 256, 256);
  hipLaunchKernelGGL(wt_transpose_f16, dim3(768),  dim3(256), 0, stream, W_c1,   wc1T,   768, 256, 768, 256);
  hipLaunchKernelGGL(wt_transpose_f16, dim3(256),  dim3(256), 0, stream, W_c2,   wc2T,   256, 256, 256, 256);
  hipLaunchKernelGGL(wt_transpose_f16, dim3(256),  dim3(256), 0, stream, W_gcn,  wgcnT,  256, 256, 256, 256);
  hipLaunchKernelGGL(wt_transpose_f16, dim3(160),  dim3(256), 0, stream, W_objp, wobjpT, 256, OCC, 256, 160);
  hipLaunchKernelGGL(wt_transpose_f16, dim3(64),   dim3(256), 0, stream, W_relp, wrelpT, 256, RCC, 256, 64);
  for (int t = 0; t < 3; ++t) {
    hipLaunchKernelGGL(wt_transpose_f16, dim3(64), dim3(256), 0, stream,
                       att1_w + (size_t)t * 256 * 64, watt1T + (size_t)t * 64 * 256, 256, 64, 256, 64);
    hipLaunchKernelGGL(wt_transpose_f16, dim3(64), dim3(256), 0, stream,
                       att2_w + (size_t)t * 64 * 256, watt2T + (size_t)t * 256 * 64, 64, 256, 64, 256);
  }

  // --- obj_in concat -> f16 ---
  hipLaunchKernelGGL(build_objin16, dim3((NN * K1P + 255) / 256), dim3(256), 0, stream,
                     roi, logits, bbox, objin16);

  // --- the big bandwidth pass: masked spatial means of union_features ---
  hipLaunchKernelGGL(masked_mean_kernel, dim3(PP), dim3(256), 0, stream, uf, bbox, pairs, sF32, s16);

  // --- object embedding MLP ---
  hipLaunchKernelGGL(gemm_wmma_f16, dim3(2, 4), wblk, 0, stream,
                     objin16, wemb1T, b_emb1, (float*)nullptr, h1o16, 256, 256, 16, K1P, 1);
  hipLaunchKernelGGL(gemm_wmma_f16, dim3(2, 4), wblk, 0, stream,
                     h1o16, wemb2T, b_emb2, feats, feats16, 256, 256, 16, 256, 0);

  // --- attention squeeze/excite per branch (sbj/obj/bg) ---
  for (int t = 0; t < 3; ++t) {
    hipLaunchKernelGGL(gemm_wmma_f16, dim3(62, 1), wblk, 0, stream,
                       s16 + (size_t)t * PP * DD, watt1T + (size_t)t * 64 * 256,
                       att1_b + (size_t)t * 64, (float*)nullptr, g1h, 64, 64, 4, 256, 1);
    hipLaunchKernelGGL(gemm_wmma_f16, dim3(62, 4), wblk, 0, stream,
                       g1h, watt2T + (size_t)t * 256 * 64,
                       att2_b + (size_t)t * 256, pre, (_Float16*)nullptr, 256, 256, 16, 64, 0);
    hipLaunchKernelGGL(gate_kernel, dim3(PP), dim3(256), 0, stream, sF32, pre, rel16, t);
  }

  // --- relation MLP ---
  hipLaunchKernelGGL(gemm_wmma_f16, dim3(62, 4), wblk, 0, stream,
                     rel16, wc1T, b_c1, (float*)nullptr, h1r16, 256, 256, 16, 768, 1);
  hipLaunchKernelGGL(gemm_wmma_f16, dim3(62, 4), wblk, 0, stream,
                     h1r16, wc2T, b_c2, feats + (size_t)NN * DD, feats16 + (size_t)NN * DD,
                     256, 256, 16, 256, 0);

  // --- GCN: h = feats @ W_gcn, then structural norm_adj combine ---
  hipLaunchKernelGGL(gemm_wmma_f16, dim3(64, 4), wblk, 0, stream,
                     feats16, wgcnT, (const float*)nullptr, hbuf, (_Float16*)nullptr,
                     256, 256, 16, 256, 0);
  hipLaunchKernelGGL(gcn_obj_kernel, dim3(NN), dim3(256), 0, stream, hbuf, feats, b_gcn, pairs, gcn16);
  hipLaunchKernelGGL(gcn_rel_kernel, dim3(PP), dim3(256), 0, stream, hbuf, feats, b_gcn, pairs, gcn16);

  // --- final projections straight into d_out ---
  hipLaunchKernelGGL(gemm_wmma_f16, dim3(2, 3), wblk, 0, stream,
                     gcn16, wobjpT, b_objp, out, (_Float16*)nullptr, OCC, OCC, 10, 256, 0);
  hipLaunchKernelGGL(gemm_wmma_f16, dim3(62, 1), wblk, 0, stream,
                     gcn16 + (size_t)NN * DD, wrelpT, b_relp, out + (size_t)NN * OCC,
                     (_Float16*)nullptr, RCC, RCC, 4, 256, 0);
}
